// Sequence_fc_21277267984827
// MI455X (gfx1250) — matrix operand order repaired and checked by reading, not compiled, not run on hardware
//
#include <hip/hip_runtime.h>
#include <hip/hip_bf16.h>

#define B_   128
#define T_   128
#define DIN  64
#define H_   1024
#define G4   4096
#define NWG  64     // persistent workgroups (H_/16 column strips)

typedef __bf16 bf16_t;
typedef __attribute__((ext_vector_type(16))) __bf16 v16bf;
typedef __attribute__((ext_vector_type(8)))  float  v8f;

union V16U { v16bf v; uint4 u[2]; };

// ---- WMMA fragment loaders (CDNA5 ISA 7.12.2 layouts, wave32) -------------
// A: 16-bit 16x32 (MxK), row-major source [rows, ld]:
//   lane<16 : M=lane,    elems 0..7 -> K=k0..k0+7,   elems 8..15 -> K=k0+16..k0+23
//   lane>=16: M=lane-16, elems 0..7 -> K=k0+8..+15,  elems 8..15 -> K=k0+24..+31
__device__ __forceinline__ v16bf load_fragA(const bf16_t* __restrict__ base, int ld,
                                            int m0, int k0, int lane) {
    const int half = lane >> 4;
    const bf16_t* p = base + (size_t)(m0 + (lane & 15)) * ld + k0 + half * 8;
    V16U r;
    r.u[0] = *(const uint4*)(p);
    r.u[1] = *(const uint4*)(p + 16);
    return r.v;
}

// B: 16-bit 32x16 (KxN) built from row-major W[n, k] (W^T on the fly):
//   lane<16 : N=lane, K=k0..k0+7,k0+16..k0+23 ; lane>=16: N=lane-16, K=k0+8..k0+15,k0+24..k0+31
__device__ __forceinline__ v16bf load_fragB(const bf16_t* __restrict__ w, int ld,
                                            int n0, int k0, int lane) {
    const bf16_t* p = w + (size_t)(n0 + (lane & 15)) * ld + k0 + ((lane >> 4) << 3);
    V16U r;
    r.u[0] = *(const uint4*)(p);
    r.u[1] = *(const uint4*)(p + 16);
    return r.v;
}

__device__ __forceinline__ float sigmoidf_(float x) { return 1.0f / (1.0f + __expf(-x)); }

// ---- device-wide split barrier (arrive + generation spin) ------------------
__device__ __forceinline__ void global_barrier(unsigned* counter, unsigned& phase) {
    __threadfence();                       // release my stores to agent scope
    __syncthreads();                       // whole WG arrived
    phase += NWG;
    if (threadIdx.x == 0) {
        __hip_atomic_fetch_add(counter, 1u, __ATOMIC_RELEASE, __HIP_MEMORY_SCOPE_AGENT);
        while (__hip_atomic_load(counter, __ATOMIC_ACQUIRE, __HIP_MEMORY_SCOPE_AGENT) < phase) {
            __builtin_amdgcn_s_sleep(2);
        }
    }
    __syncthreads();
    __threadfence();                       // acquire side
}

// ---- K-accumulation over one weight matrix ---------------------------------
__device__ __forceinline__ void gemm_acc(v8f acc[8],
                                         const bf16_t* __restrict__ A, int ldA,
                                         const bf16_t* __restrict__ W, int K,
                                         int n0, int lane) {
    for (int k0 = 0; k0 < K; k0 += 32) {
        v16bf bm = load_fragB(W, K, n0, k0, lane);
        #pragma unroll
        for (int mt = 0; mt < 8; ++mt) {
            v16bf am = load_fragA(A, ldA, mt * 16, k0, lane);
            acc[mt] = __builtin_amdgcn_wmma_f32_16x16x32_bf16(
                false, am, false, bm, (short)0, acc[mt], false, false);
        }
    }
}

// ---- one LSTM layer step for this WG's 16 hidden columns -------------------
// wave g (=threadIdx.x>>5) computes gate g's [128 x 16] pre-activation strip.
__device__ __forceinline__ void layer_step(
    const bf16_t* __restrict__ A0, int ldA0, const bf16_t* __restrict__ W0, int K0,
    const bf16_t* __restrict__ A1, const bf16_t* __restrict__ W1, int K1,
    const float* __restrict__ bias, float* __restrict__ c, bf16_t* __restrict__ h_out,
    float* gates, int jblk, int g, int lane, int tid)
{
    const int n0 = g * H_ + jblk;
    v8f acc[8] = {};
    gemm_acc(acc, A0, ldA0, W0, K0, n0, lane);
    gemm_acc(acc, A1, H_,   W1, K1, n0, lane);

    // C layout: elem r -> M = mt*16 + r + (lane<16?0:8), N = lane&15
    const float bval = bias[n0 + (lane & 15)];
    const int   colL = lane & 15;
    const int   rOff = (lane >> 4) << 3;
    #pragma unroll
    for (int mt = 0; mt < 8; ++mt)
        #pragma unroll
        for (int r = 0; r < 8; ++r)
            gates[((size_t)g * B_ + mt * 16 + rOff + r) * 16 + colL] = acc[mt][r] + bval;

    __syncthreads();

    // fused cell update: 128 rows x 16 cols over 128 threads
    for (int idx = tid; idx < B_ * 16; idx += 128) {
        int row = idx >> 4;
        int col = idx & 15;
        int j   = jblk + col;
        float iv = gates[((size_t)0 * B_ + row) * 16 + col];
        float fv = gates[((size_t)1 * B_ + row) * 16 + col];
        float gv = gates[((size_t)2 * B_ + row) * 16 + col];
        float ov = gates[((size_t)3 * B_ + row) * 16 + col];
        float co = c[row * H_ + j];
        float cn = sigmoidf_(fv) * co + sigmoidf_(iv) * tanhf(gv);
        float hn = sigmoidf_(ov) * tanhf(cn);
        c[row * H_ + j]     = cn;
        h_out[row * H_ + j] = (bf16_t)hn;
    }
}

// ---- one 16x16 tile of a linear head ---------------------------------------
__device__ __forceinline__ void linear_tile(
    const bf16_t* __restrict__ A, int ldA, const bf16_t* __restrict__ W, int K,
    const float* __restrict__ bias, int N, int m0, int n0, int lane,
    float* __restrict__ outf, bf16_t* __restrict__ outbf)
{
    v8f acc = {};
    for (int k0 = 0; k0 < K; k0 += 32) {
        v16bf bm = load_fragB(W, K, n0, k0, lane);
        v16bf am = load_fragA(A, ldA, m0, k0, lane);
        acc = __builtin_amdgcn_wmma_f32_16x16x32_bf16(
            false, am, false, bm, (short)0, acc, false, false);
    }
    const float bval = bias[n0 + (lane & 15)];
    const int   col  = n0 + (lane & 15);
    const int   rOff = (lane >> 4) << 3;
    #pragma unroll
    for (int r = 0; r < 8; ++r) {
        int row = m0 + rOff + r;
        float v = acc[r] + bval;
        if (outf)  outf[row * N + col]  = v;
        if (outbf) outbf[row * N + col] = (bf16_t)v;
    }
}

// ---- prep kernels ----------------------------------------------------------
__global__ void cvt_f32_bf16(const float* __restrict__ src, bf16_t* __restrict__ dst, int n) {
    int i = blockIdx.x * blockDim.x + threadIdx.x;
    if (i < n) dst[i] = (bf16_t)src[i];
}

__global__ void add_bias2(const float* __restrict__ a, const float* __restrict__ b,
                          float* __restrict__ dst, int n) {
    int i = blockIdx.x * blockDim.x + threadIdx.x;
    if (i < n) dst[i] = a[i] + b[i];
}

// ---- persistent LSTM: whole scan + heads + future step in ONE launch -------
// grid = 64 WGs x 128 threads; WG owns hidden columns [16*bid, 16*bid+16).
__global__ __launch_bounds__(128) void lstm_persistent(
    const bf16_t* __restrict__ xbf,
    const bf16_t* __restrict__ wih1, const bf16_t* __restrict__ whh1,
    const bf16_t* __restrict__ wih2, const bf16_t* __restrict__ whh2,
    const bf16_t* __restrict__ wlin, const bf16_t* __restrict__ whio,
    const float* __restrict__ bias1, const float* __restrict__ bias2,
    const float* __restrict__ b_lin, const float* __restrict__ b_hio,
    float* __restrict__ c1, float* __restrict__ c2,
    bf16_t* h1a, bf16_t* h1b, bf16_t* h2a, bf16_t* h2b,
    bf16_t* __restrict__ outbf, float* __restrict__ out,
    unsigned* __restrict__ barrier_cnt)
{
    __shared__ float gates[4 * B_ * 16];   // 32 KB

    const int tid  = threadIdx.x;
    const int lane = tid & 31;
    const int g    = tid >> 5;             // wave == gate
    const int jblk = blockIdx.x * 16;

    bf16_t* h1p[2] = { h1a, h1b };
    bf16_t* h2p[2] = { h2a, h2b };
    int cur1 = 0, cur2 = 0;
    unsigned phase = 0;

    // ---- recurrent scan over T ----
    for (int t = 0; t < T_; ++t) {
        layer_step(xbf + (size_t)t * DIN, T_ * DIN, wih1, DIN,
                   h1p[cur1], whh1, H_, bias1, c1, h1p[cur1 ^ 1],
                   gates, jblk, g, lane, tid);
        cur1 ^= 1;
        global_barrier(barrier_cnt, phase);

        layer_step(h1p[cur1], H_, wih2, H_,
                   h2p[cur2], whh2, H_, bias2, c2, h2p[cur2 ^ 1],
                   gates, jblk, g, lane, tid);
        cur2 ^= 1;
        global_barrier(barrier_cnt, phase);
    }

    // ---- output = h2 @ W_lin^T + b_lin (bf16; feeds the future step) ----
    if (blockIdx.x < 8)   // 8 WGs x 4 waves = 32 tiles = [128 x 64]
        linear_tile(h2p[cur2], H_, wlin, H_, b_lin, DIN,
                    blockIdx.x * 16, g * 16, lane, nullptr, outbf);
    global_barrier(barrier_cnt, phase);

    // ---- future = 1 step ----
    layer_step(outbf, DIN, wih1, DIN,
               h1p[cur1], whh1, H_, bias1, c1, h1p[cur1 ^ 1],
               gates, jblk, g, lane, tid);
    cur1 ^= 1;
    global_barrier(barrier_cnt, phase);

    layer_step(h1p[cur1], H_, wih2, H_,
               h2p[cur2], whh2, H_, bias2, c2, h2p[cur2 ^ 1],
               gates, jblk, g, lane, tid);
    cur2 ^= 1;
    global_barrier(barrier_cnt, phase);

    // ---- final: out = h2 @ W_hio^T + b_hio -> [B, 1, 64] fp32 ----
    if (blockIdx.x < 8)
        linear_tile(h2p[cur2], H_, whio, H_, b_hio, DIN,
                    blockIdx.x * 16, g * 16, lane, out, nullptr);
}

// ---------------------------------------------------------------------------
extern "C" void kernel_launch(void* const* d_in, const int* in_sizes, int n_in,
                              void* d_out, int out_size, void* d_ws, size_t ws_size,
                              hipStream_t stream) {
    const float* x     = (const float*)d_in[0];
    const float* W_ih1 = (const float*)d_in[1];
    const float* W_hh1 = (const float*)d_in[2];
    const float* b_ih1 = (const float*)d_in[3];
    const float* b_hh1 = (const float*)d_in[4];
    const float* W_ih2 = (const float*)d_in[5];
    const float* W_hh2 = (const float*)d_in[6];
    const float* b_ih2 = (const float*)d_in[7];
    const float* b_hh2 = (const float*)d_in[8];
    const float* W_lin = (const float*)d_in[9];
    const float* b_lin = (const float*)d_in[10];
    const float* W_hio = (const float*)d_in[11];
    const float* b_hio = (const float*)d_in[12];
    float* out = (float*)d_out;

    // ---- workspace carve-out (256B aligned) ----
    char*  ws  = (char*)d_ws;
    size_t off = 0;
    auto alloc = [&](size_t bytes) -> void* {
        void* p = ws + off;
        off += (bytes + 255) & ~(size_t)255;
        return p;
    };
    bf16_t*   wih1  = (bf16_t*)alloc((size_t)G4 * DIN * 2);
    bf16_t*   whh1  = (bf16_t*)alloc((size_t)G4 * H_ * 2);
    bf16_t*   wih2  = (bf16_t*)alloc((size_t)G4 * H_ * 2);
    bf16_t*   whh2  = (bf16_t*)alloc((size_t)G4 * H_ * 2);
    bf16_t*   wlin  = (bf16_t*)alloc((size_t)DIN * H_ * 2);
    bf16_t*   whio  = (bf16_t*)alloc((size_t)DIN * H_ * 2);
    bf16_t*   xbf   = (bf16_t*)alloc((size_t)B_ * T_ * DIN * 2);
    bf16_t*   h1b0  = (bf16_t*)alloc((size_t)B_ * H_ * 2);
    bf16_t*   h1b1  = (bf16_t*)alloc((size_t)B_ * H_ * 2);
    bf16_t*   h2b0  = (bf16_t*)alloc((size_t)B_ * H_ * 2);
    bf16_t*   h2b1  = (bf16_t*)alloc((size_t)B_ * H_ * 2);
    bf16_t*   outbf = (bf16_t*)alloc((size_t)B_ * DIN * 2);
    float*    bias1 = (float*)alloc((size_t)G4 * 4);
    float*    bias2 = (float*)alloc((size_t)G4 * 4);
    float*    c1    = (float*)alloc((size_t)B_ * H_ * 4);
    float*    c2    = (float*)alloc((size_t)B_ * H_ * 4);
    unsigned* bcnt  = (unsigned*)alloc(256);
    (void)ws_size; (void)in_sizes; (void)n_in; (void)out_size;

    // ---- prep: bf16 conversion, fused biases, zero state + barrier ----
    auto cvt = [&](const float* s, bf16_t* d, int n) {
        cvt_f32_bf16<<<(n + 255) / 256, 256, 0, stream>>>(s, d, n);
    };
    cvt(W_ih1, wih1, G4 * DIN);
    cvt(W_hh1, whh1, G4 * H_);
    cvt(W_ih2, wih2, G4 * H_);
    cvt(W_hh2, whh2, G4 * H_);
    cvt(W_lin, wlin, DIN * H_);
    cvt(W_hio, whio, DIN * H_);
    cvt(x,     xbf,  B_ * T_ * DIN);
    add_bias2<<<(G4 + 255) / 256, 256, 0, stream>>>(b_ih1, b_hh1, bias1, G4);
    add_bias2<<<(G4 + 255) / 256, 256, 0, stream>>>(b_ih2, b_hh2, bias2, G4);
    hipMemsetAsync(c1,   0, (size_t)B_ * H_ * 4, stream);
    hipMemsetAsync(c2,   0, (size_t)B_ * H_ * 4, stream);
    hipMemsetAsync(h1b0, 0, (size_t)B_ * H_ * 2, stream);
    hipMemsetAsync(h2b0, 0, (size_t)B_ * H_ * 2, stream);
    hipMemsetAsync(bcnt, 0, 256, stream);

    // ---- single persistent kernel: scan + head + future + final head ----
    lstm_persistent<<<NWG, 128, 0, stream>>>(
        xbf, wih1, whh1, wih2, whh2, wlin, whio,
        bias1, bias2, b_lin, b_hio,
        c1, c2, h1b0, h1b1, h2b0, h2b1,
        outbf, out, bcnt);
}
